// HybridHebbianLayer_39281770889410
// MI455X (gfx1250) — matrix operand order repaired and checked by reading, not compiled, not run on hardware
//
#include <hip/hip_runtime.h>

// ---------------------------------------------------------------------------
// Types / helpers
// ---------------------------------------------------------------------------
typedef __bf16 bf16_t;
typedef __attribute__((ext_vector_type(16))) __bf16 v16bf;
typedef __attribute__((ext_vector_type(8)))  float  v8f;
typedef __attribute__((ext_vector_type(4)))  unsigned int v4u;
typedef __attribute__((ext_vector_type(8)))  int v8i;
typedef __attribute__((ext_vector_type(4)))  int v4i;

#define D_MODEL 1024
#define DI_DIM  2048
#define NTOK    16384      // B*S
#define SEQ     8192
#define CHUNK   64
#define NCHUNK  128
#define ALPHA_F 0.9f

__device__ __forceinline__ bf16_t f2bf(float f) {
  unsigned u = __builtin_bit_cast(unsigned, f);
  unsigned r = u + 0x7FFFu + ((u >> 16) & 1u);
  unsigned short h = (unsigned short)(r >> 16);
  return __builtin_bit_cast(bf16_t, h);
}

union FragU { v16bf v; uint4 q[2]; bf16_t h[16]; };

__device__ __forceinline__ v8f wmma_bf16(v16bf a, v16bf b, v8f c) {
  return __builtin_amdgcn_wmma_f32_16x16x32_bf16(false, a, false, b, (short)0, c,
                                                 false, false);
}

// A operand 16x32 from row-major source (ld elems/row), tile at (row0, kb).
__device__ __forceinline__ v16bf load_A_rowmajor(const bf16_t* A, int ld,
                                                 int row0, int kb, int lane) {
  int l15 = lane & 15, g = lane >> 4;
  const bf16_t* p = A + (size_t)(row0 + l15) * ld + kb + g * 16;
  FragU u;
  u.q[0] = *(const uint4*)(p);
  u.q[1] = *(const uint4*)(p + 8);
  return u.v;
}

// B operand 32x16 from an N-major (transposed) source: b[j]=BT[(n0+l15)*ld+kb+g*16+j]
__device__ __forceinline__ v16bf load_B_ntrans(const bf16_t* BT, int ld,
                                               int n0, int kb, int lane) {
  int l15 = lane & 15, g = lane >> 4;
  const bf16_t* p = BT + (size_t)(n0 + l15) * ld + kb + g * 16;
  FragU u;
  u.q[0] = *(const uint4*)(p);
  u.q[1] = *(const uint4*)(p + 8);
  return u.v;
}

// B operand 32x16 directly from row-major [K x N] source (strided gather).
__device__ __forceinline__ v16bf load_B_rowmajor(const bf16_t* B, int ld,
                                                 int kb, int n0, int lane) {
  int l15 = lane & 15, g = lane >> 4;
  FragU u;
  const bf16_t* p = B + (size_t)(kb + g * 16) * ld + n0 + l15;
#pragma unroll
  for (int j = 0; j < 16; ++j) u.h[j] = p[(size_t)j * ld];
  return u.v;
}

// A operand 16x32 where logical A[m,k] = S[k,m] (S row-major, ld elems/row).
__device__ __forceinline__ v16bf load_A_coltrans(const bf16_t* S, int ld,
                                                 int row0, int kb, int lane) {
  int l15 = lane & 15, g = lane >> 4;
  FragU u;
  const bf16_t* p = S + (size_t)(kb + g * 16) * ld + row0 + l15;
#pragma unroll
  for (int j = 0; j < 8; ++j) u.h[j] = p[(size_t)j * ld];
  const bf16_t* p2 = p + (size_t)8 * ld;
#pragma unroll
  for (int j = 0; j < 8; ++j) u.h[8 + j] = p2[(size_t)j * ld];
  return u.v;
}

// ---------------------------------------------------------------------------
// Tensor Data Mover: 2D tile (tile_rows x tile_cols bf16) from a row-major
// tensor with row pitch K_elems into LDS at byte offset lds_off (contiguous).
// ---------------------------------------------------------------------------
__device__ __forceinline__ void tdm_load_2d(unsigned lds_off, const void* gptr,
                                            unsigned K_elems, unsigned tile_rows,
                                            unsigned tile_cols) {
  unsigned long long ga = (unsigned long long)gptr;
  v4u g0;
  g0[0] = 1u;                                   // count=1, user descriptor
  g0[1] = lds_off;                              // lds_addr (bytes)
  g0[2] = (unsigned)ga;                         // global_addr lo
  g0[3] = (unsigned)(ga >> 32) | (2u << 30);    // global_addr hi | type=2
  v8i g1;
  g1[0] = (int)(1u << 16);                      // workgroup_mask=0, data_size=2B
  g1[1] = (int)((K_elems & 0xFFFFu) << 16);     // tensor_dim0 lo
  g1[2] = (int)(K_elems >> 16);                 // tensor_dim0 hi, tensor_dim1 lo=0
  g1[3] = (int)(16u | (tile_cols << 16));       // tensor_dim1 hi (=1<<20), tile_dim0
  g1[4] = (int)tile_rows;                       // tile_dim1, tile_dim2=0
  g1[5] = (int)K_elems;                         // tensor_dim0_stride lo
  g1[6] = 0;                                    // stride0 hi, stride1 lo
  g1[7] = 0;                                    // stride1 hi
  v4i g2 = {1, 0, 0, 0};                        // tensor_dim2=1 (unused)
  v4i g3 = {0, 0, 0, 0};
#if defined(__clang_major__) && (__clang_major__ >= 23)
  v8i g4 = {0, 0, 0, 0, 0, 0, 0, 0};
  __builtin_amdgcn_tensor_load_to_lds(g0, g1, g2, g3, g4, 0);
#else
  __builtin_amdgcn_tensor_load_to_lds(g0, g1, g2, g3, 0);
#endif
}

// ---------------------------------------------------------------------------
// fp32 -> bf16 conversion with transpose: s [K x N] -> d [N x K]
// ---------------------------------------------------------------------------
__global__ __launch_bounds__(256) void cvt_transpose_bf16(
    const float* __restrict__ s, bf16_t* __restrict__ d, int K, int N) {
  int idx = blockIdx.x * 256 + threadIdx.x;
  int k = idx / N, n = idx - k * N;
  d[(size_t)n * K + k] = f2bf(s[idx]);
}

// ---------------------------------------------------------------------------
// RMSNorm -> bf16
// ---------------------------------------------------------------------------
__global__ __launch_bounds__(256) void rmsnorm_bf16(const float* __restrict__ x,
                                                    const float* __restrict__ w,
                                                    bf16_t* __restrict__ out) {
  __shared__ float sbuf[256];
  int tid = threadIdx.x;
  int row = blockIdx.x;
  const float* xr = x + (size_t)row * D_MODEL;
  float ss = 0.f;
#pragma unroll
  for (int i = tid; i < D_MODEL; i += 256) { float v = xr[i]; ss += v * v; }
  sbuf[tid] = ss;
  __syncthreads();
  for (int s = 128; s > 0; s >>= 1) {
    if (tid < s) sbuf[tid] += sbuf[tid + s];
    __syncthreads();
  }
  float rs = rsqrtf(sbuf[0] * (1.0f / D_MODEL) + 1e-5f);
#pragma unroll
  for (int i = tid; i < D_MODEL; i += 256)
    out[(size_t)row * D_MODEL + i] = f2bf(xr[i] * rs * w[i]);
}

// ---------------------------------------------------------------------------
// TDM double-buffered bf16 WMMA GEMM: C = scale * (A @ BT^T) [+ add]
// A:  [M x K] row-major bf16
// BT: [N x K] row-major bf16 (i.e. B transposed)
// BM=BN=128, BK=32; 256 threads = 8 waves (2x4), 64x32 tile per wave.
// Wave 0 issues two TDM descriptors per K-step into the back buffer while
// all waves run WMMA on the front buffer; tensorcnt + barrier publishes it.
// ---------------------------------------------------------------------------
template <bool WF, bool WBF, bool ADD>
__global__ __launch_bounds__(256) void gemm_tdm(
    const bf16_t* __restrict__ A, const bf16_t* __restrict__ BT,
    float* __restrict__ Cf, bf16_t* __restrict__ Cbf,
    const float* __restrict__ addsrc, int M, int N, int K, float scale) {
  __shared__ __align__(16) bf16_t As[2][128][32];
  __shared__ __align__(16) bf16_t Bs[2][128][32];
  int tid = threadIdx.x;
  int lane = tid & 31, wid = tid >> 5;
  int wm = wid >> 2, wn = wid & 3;
  int row0 = blockIdx.x * 128;
  int n0 = blockIdx.y * 128;
  bool issuer = (__builtin_amdgcn_readfirstlane(wid) == 0);

  unsigned ldsA[2] = {(unsigned)(unsigned long long)(void*)&As[0][0][0],
                      (unsigned)(unsigned long long)(void*)&As[1][0][0]};
  unsigned ldsB[2] = {(unsigned)(unsigned long long)(void*)&Bs[0][0][0],
                      (unsigned)(unsigned long long)(void*)&Bs[1][0][0]};

  v8f acc[4][2];
#pragma unroll
  for (int mt = 0; mt < 4; ++mt)
#pragma unroll
    for (int nt = 0; nt < 2; ++nt)
#pragma unroll
      for (int e = 0; e < 8; ++e) acc[mt][nt][e] = 0.f;

  int nk = K >> 5;
  if (issuer) {
    tdm_load_2d(ldsA[0], A + (size_t)row0 * K, K, 128, 32);
    tdm_load_2d(ldsB[0], BT + (size_t)n0 * K, K, 128, 32);
    __builtin_amdgcn_s_wait_tensorcnt((short)0);
  }
  __syncthreads();

  for (int i = 0; i < nk; ++i) {
    int b = i & 1;
    if (issuer && (i + 1) < nk) {
      int k1 = (i + 1) * 32;
      tdm_load_2d(ldsA[b ^ 1], A + (size_t)row0 * K + k1, K, 128, 32);
      tdm_load_2d(ldsB[b ^ 1], BT + (size_t)n0 * K + k1, K, 128, 32);
    }

    v16bf af[4], bfg[2];
#pragma unroll
    for (int mt = 0; mt < 4; ++mt)
      af[mt] = load_A_rowmajor(&As[b][0][0], 32, wm * 64 + mt * 16, 0, lane);
#pragma unroll
    for (int nt = 0; nt < 2; ++nt)
      bfg[nt] = load_B_ntrans(&Bs[b][0][0], 32, wn * 32 + nt * 16, 0, lane);
#pragma unroll
    for (int mt = 0; mt < 4; ++mt)
#pragma unroll
      for (int nt = 0; nt < 2; ++nt)
        acc[mt][nt] = wmma_bf16(af[mt], bfg[nt], acc[mt][nt]);

    if (issuer && (i + 1) < nk) __builtin_amdgcn_s_wait_tensorcnt((short)0);
    __syncthreads();
  }

  int l15 = lane & 15, g = lane >> 4;
#pragma unroll
  for (int mt = 0; mt < 4; ++mt)
#pragma unroll
    for (int nt = 0; nt < 2; ++nt)
#pragma unroll
      for (int e = 0; e < 8; ++e) {
        int row = row0 + wm * 64 + mt * 16 + e + g * 8;
        int col = n0 + wn * 32 + nt * 16 + l15;
        float vv = acc[mt][nt][e] * scale;
        size_t idx = (size_t)row * N + col;
        if constexpr (ADD) vv += addsrc[idx];
        if constexpr (WF) Cf[idx] = vv;
        if constexpr (WBF) Cbf[idx] = f2bf(vv);
      }
}

// ---------------------------------------------------------------------------
// Fused causal depthwise conv (K=4) + SiLU + gate*SiLU -> bf16
// ---------------------------------------------------------------------------
__global__ __launch_bounds__(256) void conv_gate_bf16(
    const float* __restrict__ u, const float* __restrict__ gpre,
    const float* __restrict__ cw, const float* __restrict__ cb,
    bf16_t* __restrict__ y) {
  size_t idx = (size_t)blockIdx.x * 256 + threadIdx.x;  // over NTOK*DI
  int c = (int)(idx & (DI_DIM - 1));
  size_t t = idx >> 11;                // token
  int s = (int)(t & (SEQ - 1));        // pos within sequence
  float acc = cb[c];
#pragma unroll
  for (int kk = 0; kk < 4; ++kk) {
    int ds = s - 3 + kk;
    if (ds >= 0)
      acc += u[(t - 3 + kk) * DI_DIM + c] * cw[kk * DI_DIM + c];
  }
  float val = acc / (1.f + __expf(-acc));
  float gp = gpre[idx];
  float gate = gp / (1.f + __expf(-gp));
  y[idx] = f2bf(gate * val);
}

// ---------------------------------------------------------------------------
// Chunked Hebbian memory (see round-1 notes). WG owns (batch, 32-col block of M);
// M kept in WMMA f32 accumulators, bf16^T mirror in LDS for the q@M term.
// ---------------------------------------------------------------------------
__global__ __launch_bounds__(256) void hebbian_scan(
    const bf16_t* __restrict__ q, const bf16_t* __restrict__ k,
    const bf16_t* __restrict__ v, bf16_t* __restrict__ o) {
  extern __shared__ __align__(16) char hsm[];
  bf16_t* MbfT = (bf16_t*)hsm;                        // [32 cols][1024 dd]
  bf16_t* attn = (bf16_t*)(hsm + 32 * 1024 * 2);      // [8 waves][16][64]

  int tid = threadIdx.x, lane = tid & 31, wid = tid >> 5;
  int l15 = lane & 15, g = lane >> 4;
  int nb = blockIdx.x;
  int bb = blockIdx.y;
  int mt = wid & 3, nt = wid >> 2;
  size_t base = (size_t)bb * SEQ * D_MODEL;
  int colbase = nb * 32;

  for (int i = tid; i < 32 * 1024 / 2; i += 256) ((unsigned int*)MbfT)[i] = 0u;
  v8f Macc[8][2];
#pragma unroll
  for (int a = 0; a < 8; ++a)
#pragma unroll
    for (int b2 = 0; b2 < 2; ++b2)
#pragma unroll
      for (int e = 0; e < 8; ++e) Macc[a][b2][e] = 0.f;
  __syncthreads();

  for (int ci = 0; ci < NCHUNK; ++ci) {
    const bf16_t* qc = q + base + (size_t)ci * CHUNK * D_MODEL;
    const bf16_t* kc = k + base + (size_t)ci * CHUNK * D_MODEL;
    const bf16_t* vc = v + base + (size_t)ci * CHUNK * D_MODEL;

    // intra-chunk attention rows (q@k^T, K=1024)
    v8f att[4];
#pragma unroll
    for (int ct = 0; ct < 4; ++ct)
#pragma unroll
      for (int e = 0; e < 8; ++e) att[ct][e] = 0.f;
    for (int kb = 0; kb < D_MODEL; kb += 32) {
      v16bf aq = load_A_rowmajor(qc, D_MODEL, mt * 16, kb, lane);
#pragma unroll
      for (int ct = 0; ct < 4; ++ct) {
        v16bf bk = load_B_ntrans(kc, D_MODEL, ct * 16, kb, lane);
        att[ct] = wmma_bf16(aq, bk, att[ct]);
      }
    }
    // causal mask + stage bf16 into per-wave LDS region
    bf16_t* aw = attn + wid * 16 * 64;
#pragma unroll
    for (int ct = 0; ct < 4; ++ct)
#pragma unroll
      for (int e = 0; e < 8; ++e) {
        int irow = e + g * 8;
        int i = mt * 16 + irow;
        int j = ct * 16 + l15;
        float xv = (j <= i) ? att[ct][e] : 0.f;
        aw[irow * 64 + j] = f2bf(xv);
      }

    // o tile: inter (q@M, K=1024) + intra (attn@v, K=64)
    v8f oacc;
#pragma unroll
    for (int e = 0; e < 8; ++e) oacc[e] = 0.f;
    for (int kb = 0; kb < D_MODEL; kb += 32) {
      v16bf aq = load_A_rowmajor(qc, D_MODEL, mt * 16, kb, lane);
      v16bf bM = load_B_ntrans(MbfT, 1024, nt * 16, kb, lane);
      oacc = wmma_bf16(aq, bM, oacc);
    }
#pragma unroll
    for (int kb = 0; kb < CHUNK; kb += 32) {
      v16bf aa = load_A_rowmajor(aw, 64, 0, kb, lane);
      v16bf bv = load_B_rowmajor(vc, D_MODEL, kb, colbase + nt * 16, lane);
      oacc = wmma_bf16(aa, bv, oacc);
    }
#pragma unroll
    for (int e = 0; e < 8; ++e) {
      int row = ci * CHUNK + mt * 16 + e + g * 8;
      int col = colbase + nt * 16 + l15;
      o[base + (size_t)row * D_MODEL + col] = f2bf(oacc[e]);
    }

    // state update: M = alpha*M + k^T @ v
#pragma unroll
    for (int a = 0; a < 8; ++a)
#pragma unroll
      for (int b2 = 0; b2 < 2; ++b2)
#pragma unroll
        for (int e = 0; e < 8; ++e) Macc[a][b2][e] *= ALPHA_F;
#pragma unroll
    for (int kb = 0; kb < CHUNK; kb += 32) {
      v16bf bv2[2];
#pragma unroll
      for (int ntm = 0; ntm < 2; ++ntm)
        bv2[ntm] = load_B_rowmajor(vc, D_MODEL, kb, colbase + ntm * 16, lane);
#pragma unroll
      for (int mtm = 0; mtm < 8; ++mtm) {
        v16bf ak = load_A_coltrans(kc, D_MODEL, wid * 128 + mtm * 16, kb, lane);
#pragma unroll
        for (int ntm = 0; ntm < 2; ++ntm)
          Macc[mtm][ntm] = wmma_bf16(ak, bv2[ntm], Macc[mtm][ntm]);
      }
    }

    __syncthreads();
#pragma unroll
    for (int mtm = 0; mtm < 8; ++mtm)
#pragma unroll
      for (int ntm = 0; ntm < 2; ++ntm)
#pragma unroll
        for (int e = 0; e < 8; ++e) {
          int dd = wid * 128 + mtm * 16 + e + g * 8;
          int col = ntm * 16 + l15;
          MbfT[(size_t)col * 1024 + dd] = f2bf(Macc[mtm][ntm][e]);
        }
    __syncthreads();
  }
}

// ---------------------------------------------------------------------------
// Host-side orchestration
// ---------------------------------------------------------------------------
extern "C" void kernel_launch(void* const* d_in, const int* in_sizes, int n_in,
                              void* d_out, int out_size, void* d_ws, size_t ws_size,
                              hipStream_t stream) {
  (void)in_sizes; (void)n_in; (void)out_size; (void)ws_size;
  const float* x      = (const float*)d_in[0];
  const float* norm_w = (const float*)d_in[1];
  const float* w_up   = (const float*)d_in[2];
  const float* w_gate = (const float*)d_in[3];
  const float* w_down = (const float*)d_in[4];
  const float* conv_w = (const float*)d_in[5];
  const float* conv_b = (const float*)d_in[6];
  const float* wq     = (const float*)d_in[7];
  const float* wk     = (const float*)d_in[8];
  const float* wv     = (const float*)d_in[9];
  const float* wo     = (const float*)d_in[10];
  float* out = (float*)d_out;

  // workspace layout (bytes); all weight bf16 copies are TRANSPOSED [N x K]
  char* ws = (char*)d_ws;
  size_t off = 0;
  bf16_t* normed = (bf16_t*)(ws + off); off += (size_t)NTOK * D_MODEL * 2;
  bf16_t* wupT   = (bf16_t*)(ws + off); off += (size_t)D_MODEL * DI_DIM * 2;
  bf16_t* wgateT = (bf16_t*)(ws + off); off += (size_t)D_MODEL * DI_DIM * 2;
  bf16_t* wdownT = (bf16_t*)(ws + off); off += (size_t)DI_DIM * D_MODEL * 2;
  bf16_t* wqT    = (bf16_t*)(ws + off); off += (size_t)D_MODEL * D_MODEL * 2;
  bf16_t* wkT    = (bf16_t*)(ws + off); off += (size_t)D_MODEL * D_MODEL * 2;
  bf16_t* wvT    = (bf16_t*)(ws + off); off += (size_t)D_MODEL * D_MODEL * 2;
  bf16_t* woT    = (bf16_t*)(ws + off); off += (size_t)D_MODEL * D_MODEL * 2;
  bf16_t* ybf    = (bf16_t*)(ws + off); off += (size_t)NTOK * DI_DIM * 2;
  char* big = ws + off;  // aliased region across phases
  float* u_f32 = (float*)big;
  float* g_f32 = (float*)(big + (size_t)NTOK * DI_DIM * 4);
  bf16_t* hbf = (bf16_t*)big;
  bf16_t* qbf = (bf16_t*)(big + 1 * (size_t)NTOK * D_MODEL * 2);
  bf16_t* kbf = (bf16_t*)(big + 2 * (size_t)NTOK * D_MODEL * 2);
  bf16_t* vbf = (bf16_t*)(big + 3 * (size_t)NTOK * D_MODEL * 2);
  bf16_t* obf = (bf16_t*)(big + 4 * (size_t)NTOK * D_MODEL * 2);

  // 1) weights -> transposed bf16
  {
    int n1 = D_MODEL * DI_DIM, n2 = D_MODEL * D_MODEL;
    cvt_transpose_bf16<<<n1 / 256, 256, 0, stream>>>(w_up,   wupT,   D_MODEL, DI_DIM);
    cvt_transpose_bf16<<<n1 / 256, 256, 0, stream>>>(w_gate, wgateT, D_MODEL, DI_DIM);
    cvt_transpose_bf16<<<n1 / 256, 256, 0, stream>>>(w_down, wdownT, DI_DIM, D_MODEL);
    cvt_transpose_bf16<<<n2 / 256, 256, 0, stream>>>(wq, wqT, D_MODEL, D_MODEL);
    cvt_transpose_bf16<<<n2 / 256, 256, 0, stream>>>(wk, wkT, D_MODEL, D_MODEL);
    cvt_transpose_bf16<<<n2 / 256, 256, 0, stream>>>(wv, wvT, D_MODEL, D_MODEL);
    cvt_transpose_bf16<<<n2 / 256, 256, 0, stream>>>(wo, woT, D_MODEL, D_MODEL);
  }

  // 2) RMSNorm
  rmsnorm_bf16<<<NTOK, 256, 0, stream>>>(x, norm_w, normed);

  // 3) up / gate projections (fp32 outputs)
  gemm_tdm<true, false, false><<<dim3(NTOK / 128, DI_DIM / 128), 256, 0, stream>>>(
      normed, wupT, u_f32, nullptr, nullptr, NTOK, DI_DIM, D_MODEL, 1.0f);
  gemm_tdm<true, false, false><<<dim3(NTOK / 128, DI_DIM / 128), 256, 0, stream>>>(
      normed, wgateT, g_f32, nullptr, nullptr, NTOK, DI_DIM, D_MODEL, 1.0f);

  // 4) conv + SiLU + gating -> bf16
  {
    size_t tot = (size_t)NTOK * DI_DIM;
    conv_gate_bf16<<<(unsigned)(tot / 256), 256, 0, stream>>>(u_f32, g_f32,
                                                              conv_w, conv_b, ybf);
  }

  // 5) down projection -> h (bf16)
  gemm_tdm<false, true, false><<<dim3(NTOK / 128, D_MODEL / 128), 256, 0, stream>>>(
      ybf, wdownT, nullptr, hbf, nullptr, NTOK, D_MODEL, DI_DIM, 1.0f);

  // 6) q/k/v projections (q pre-scaled by d^-0.5 = 1/32)
  gemm_tdm<false, true, false><<<dim3(NTOK / 128, D_MODEL / 128), 256, 0, stream>>>(
      hbf, wqT, nullptr, qbf, nullptr, NTOK, D_MODEL, D_MODEL, 0.03125f);
  gemm_tdm<false, true, false><<<dim3(NTOK / 128, D_MODEL / 128), 256, 0, stream>>>(
      hbf, wkT, nullptr, kbf, nullptr, NTOK, D_MODEL, D_MODEL, 1.0f);
  gemm_tdm<false, true, false><<<dim3(NTOK / 128, D_MODEL / 128), 256, 0, stream>>>(
      hbf, wvT, nullptr, vbf, nullptr, NTOK, D_MODEL, D_MODEL, 1.0f);

  // 7) Hebbian chunked scan (64 WGs, 80KB dynamic LDS each)
  {
    size_t shbytes = 32 * 1024 * 2 + 8 * 16 * 64 * 2;  // 81920
    hebbian_scan<<<dim3(32, 2), 256, shbytes, stream>>>(qbf, kbf, vbf, obf);
  }

  // 8) output projection + residual -> d_out (fp32)
  gemm_tdm<true, false, true><<<dim3(NTOK / 128, D_MODEL / 128), 256, 0, stream>>>(
      obf, woT, out, nullptr, x, NTOK, D_MODEL, D_MODEL, 1.0f);
}
